// StructureModule_38345468018990
// MI455X (gfx1250) — compile-verified, matrix operand order checked by reading, not run on hardware
//
#include <hip/hip_runtime.h>
#include <math.h>

typedef __bf16 bf16;
typedef __attribute__((ext_vector_type(16))) __bf16 bf16x16;
typedef __attribute__((ext_vector_type(8)))  __bf16 bf16x8;
typedef __attribute__((ext_vector_type(8)))  float  f32x8;

#define NRES  768
#define CS    384
#define CZ    128
#define NH    12
#define CIPA  16
#define NPQ   4
#define NPV   8
#define CATD  2112
#define NBLOCKS 8

__device__ __forceinline__ float softplusf(float x) {
  return (x > 20.f) ? x : log1pf(expf(x));
}

// ---------------------------------------------------------------------------
// Fragment load (fast): p already points at row*ld + kb; two b128 loads.
// Layout per ISA 7.12.2 (16-bit A/B, wave32): lane l holds row (l&15),
// K = kb..kb+7 (elems 0-7) and K = 16+kb..+7 (elems 8-15), kb=(l>>4)*8.
// ---------------------------------------------------------------------------
__device__ __forceinline__ bf16x16 frag_ld(const bf16* __restrict__ p, int k0)
{
  bf16x8 lo = *(const bf16x8*)(p + k0);
  bf16x8 hi = *(const bf16x8*)(p + k0 + 16);
  return __builtin_shufflevector(lo, hi, 0, 1, 2, 3, 4, 5, 6, 7,
                                 8, 9, 10, 11, 12, 13, 14, 15);
}

__device__ __forceinline__ void store_tile(
    const f32x8& acc, int mt, int nt, int lane,
    const float* __restrict__ bias, int biasRow,
    const float* __restrict__ Res, int ldr,
    float* __restrict__ Cf, int ldcf, size_t cfo,
    bf16* __restrict__ Cb, int ldcb, size_t cbo, int relu)
{
  const int n = nt + (lane & 15);
#pragma unroll
  for (int r = 0; r < 8; ++r) {
    const int m = mt + r + 8 * (lane >> 4);
    float v = acc[r];
    if (bias) v += biasRow ? bias[m] : bias[n];
    if (Res)  v += Res[(size_t)m * ldr + n];
    if (relu) v = v > 0.f ? v : 0.f;
    if (Cf) Cf[cfo + (size_t)m * ldcf + n] = v;
    if (Cb) Cb[cbo + (size_t)m * ldcb + n] = (bf16)v;
  }
}

// ---------------------------------------------------------------------------
// FAST batched WMMA GEMM (NT): C[b] = relu(A[b] @ B[b]^T + bias + Res)
// Requires M%16==0, N%16==0, K%32==0, lda/ldb%8==0, 16B-aligned bases.
// Block = 4 waves stacked along M (block tile 128x32); each wave computes a
// 32x32 tile = 2x2 WMMA subtiles with operand reuse (2 b128 loads per wmma).
// No per-element guards anywhere; subtile validity is wave-uniform.
// ---------------------------------------------------------------------------
__global__ void wmma_gemm_fast_kernel(
    const bf16* __restrict__ A, int lda, unsigned long long sA,
    const bf16* __restrict__ B, int ldb, unsigned long long sB,
    const float* __restrict__ bias, int biasRow,
    const float* __restrict__ Res, int ldr,
    float* __restrict__ Cf, int ldcf, unsigned long long sCf,
    bf16* __restrict__ Cb, int ldcb, unsigned long long sCb,
    int M, int N, int K, int reluOut)
{
  const int wave = threadIdx.x >> 5;
  const int lane = threadIdx.x & 31;
  const int m0 = blockIdx.y * 128 + wave * 32;
  const int n0 = blockIdx.x * 32;
  if (m0 >= M) return;                    // wave-uniform
  const int b = blockIdx.z;
  A += (size_t)b * sA;
  B += (size_t)b * sB;

  const int l15 = lane & 15;
  const int kb  = (lane >> 4) * 8;
  const bool vm1 = (m0 + 32) <= M;
  const bool vn1 = (n0 + 32) <= N;
  const int mr0 = m0 + l15;
  const int mr1 = vm1 ? mr0 + 16 : mr0;   // clamp if subtile absent
  const int nr0 = n0 + l15;
  const int nr1 = vn1 ? nr0 + 16 : nr0;
  const bf16* a0 = A + (size_t)mr0 * lda + kb;
  const bf16* a1 = A + (size_t)mr1 * lda + kb;
  const bf16* b0 = B + (size_t)nr0 * ldb + kb;
  const bf16* b1 = B + (size_t)nr1 * ldb + kb;

  f32x8 acc00 = {0.f,0.f,0.f,0.f,0.f,0.f,0.f,0.f};
  f32x8 acc01 = acc00, acc10 = acc00, acc11 = acc00;
  for (int k0 = 0; k0 < K; k0 += 32) {
    bf16x16 af0 = frag_ld(a0, k0);
    bf16x16 af1 = frag_ld(a1, k0);
    bf16x16 bf0 = frag_ld(b0, k0);
    bf16x16 bf1 = frag_ld(b1, k0);
    acc00 = __builtin_amdgcn_wmma_f32_16x16x32_bf16(false, af0, false, bf0, (short)0, acc00, false, false);
    acc01 = __builtin_amdgcn_wmma_f32_16x16x32_bf16(false, af0, false, bf1, (short)0, acc01, false, false);
    acc10 = __builtin_amdgcn_wmma_f32_16x16x32_bf16(false, af1, false, bf0, (short)0, acc10, false, false);
    acc11 = __builtin_amdgcn_wmma_f32_16x16x32_bf16(false, af1, false, bf1, (short)0, acc11, false, false);
  }

  const size_t cfo = (size_t)b * sCf;
  const size_t cbo = (size_t)b * sCb;
  store_tile(acc00, m0, n0, lane, bias, biasRow, Res, ldr, Cf, ldcf, cfo, Cb, ldcb, cbo, reluOut);
  if (vn1)
    store_tile(acc01, m0, n0 + 16, lane, bias, biasRow, Res, ldr, Cf, ldcf, cfo, Cb, ldcb, cbo, reluOut);
  if (vm1)
    store_tile(acc10, m0 + 16, n0, lane, bias, biasRow, Res, ldr, Cf, ldcf, cfo, Cb, ldcb, cbo, reluOut);
  if (vm1 && vn1)
    store_tile(acc11, m0 + 16, n0 + 16, lane, bias, biasRow, Res, ldr, Cf, ldcf, cfo, Cb, ldcb, cbo, reluOut);
}

// ---------------------------------------------------------------------------
// EDGE WMMA GEMM (NT, scalar-guarded) — only for tiny N (6 / 14) heads.
// ---------------------------------------------------------------------------
__global__ void wmma_gemm_edge_kernel(
    const bf16* __restrict__ A, int lda,
    const bf16* __restrict__ B, int ldb,
    const float* __restrict__ bias,
    const float* __restrict__ Res, int ldr,
    float* __restrict__ Cf, int ldcf,
    int M, int N, int K)
{
  const int wave = threadIdx.x >> 5;
  const int lane = threadIdx.x & 31;
  const int m0 = (blockIdx.y * 4 + wave) * 16;
  if (m0 >= M) return;
  const int n0 = blockIdx.x * 16;
  const int mRow = m0 + (lane & 15);
  const int nCol = n0 + (lane & 15);
  const int kb   = (lane >> 4) * 8;
  const bool mOk = (mRow < M);
  const bool nOk = (nCol < N);

  f32x8 acc = {0.f,0.f,0.f,0.f,0.f,0.f,0.f,0.f};
  for (int k0 = 0; k0 < K; k0 += 32) {
    bf16x16 af, bfv;
#pragma unroll
    for (int e = 0; e < 16; ++e) {
      const int k = k0 + ((e < 8) ? (kb + e) : (16 + kb + (e - 8)));
      af[e]  = (mOk && k < K) ? A[(size_t)mRow * lda + k] : (bf16)0.f;
      bfv[e] = (nOk && k < K) ? B[(size_t)nCol * ldb + k] : (bf16)0.f;
    }
    acc = __builtin_amdgcn_wmma_f32_16x16x32_bf16(false, af, false, bfv,
                                                  (short)0, acc, false, false);
  }
  if (nOk) {
#pragma unroll
    for (int r = 0; r < 8; ++r) {
      const int m = m0 + r + 8 * (lane >> 4);
      if (m < M) {
        float v = acc[r];
        if (bias) v += bias[nCol];
        if (Res)  v += Res[(size_t)m * ldr + nCol];
        Cf[(size_t)m * ldcf + nCol] = v;
      }
    }
  }
}

// ---------------------------------------------------------------------------
__global__ void cvt_kernel(const float* __restrict__ in, bf16* __restrict__ out,
                           int n, int relu)
{
  int i = blockIdx.x * blockDim.x + threadIdx.x;
  if (i < n) {
    float v = in[i];
    if (relu) v = v > 0.f ? v : 0.f;
    out[i] = (bf16)v;
  }
}

// Weight transpose: W (K x N fp32) -> WT (N x K bf16)
__global__ void transpose_w_kernel(const float* __restrict__ W,
                                   bf16* __restrict__ WT, int K, int N)
{
  int i = blockIdx.x * blockDim.x + threadIdx.x;
  if (i >= K * N) return;
  int n = i / K, k = i % K;
  WT[(size_t)n * K + k] = (bf16)W[(size_t)k * N + n];
}

__global__ void zero_bf16_kernel(bf16* __restrict__ p, int n)
{
  int i = blockIdx.x * blockDim.x + threadIdx.x;
  if (i < n) p[i] = (bf16)0.f;
}

// Zero pad rows (24..31 per head) of vptsT [h][32][NRES]
__global__ void zero_vptst_pad_kernel(bf16* __restrict__ vptsT)
{
  int i = blockIdx.x * blockDim.x + threadIdx.x;
  if (i >= NH * 8 * NRES) return;
  int h = i / (8 * NRES);
  int r = i % (8 * NRES);
  vptsT[((size_t)h * 32 + 24) * NRES + r] = (bf16)0.f;
}

// Pad bias vector to 16 floats
__global__ void pad_bias_kernel(const float* __restrict__ in,
                                float* __restrict__ out, int n, int total)
{
  int i = blockIdx.x * blockDim.x + threadIdx.x;
  if (i < total) out[i] = (i < n) ? in[i] : 0.f;
}

// ---------------------------------------------------------------------------
// Row LayerNorm (generic). One 128-thread block per row. In-place safe.
// ---------------------------------------------------------------------------
__global__ void layernorm_kernel(const float* __restrict__ x,
                                 const float* __restrict__ g,
                                 const float* __restrict__ bb,
                                 float* __restrict__ yf,
                                 bf16* __restrict__ yb,
                                 int dim)
{
  __shared__ float red[10];
  const int row = blockIdx.x;
  const float* xr = x + (size_t)row * dim;
  float s = 0.f, s2 = 0.f;
  for (int j = threadIdx.x; j < dim; j += blockDim.x) {
    float v = xr[j]; s += v; s2 += v * v;
  }
  for (int o = 16; o; o >>= 1) { s += __shfl_xor(s, o, 32); s2 += __shfl_xor(s2, o, 32); }
  if ((threadIdx.x & 31) == 0) { red[threadIdx.x >> 5] = s; red[4 + (threadIdx.x >> 5)] = s2; }
  __syncthreads();
  if (threadIdx.x == 0) {
    float a = 0.f, b2 = 0.f;
    const int nw = blockDim.x >> 5;
    for (int w = 0; w < nw; ++w) { a += red[w]; b2 += red[4 + w]; }
    const float mean = a / dim;
    const float var  = b2 / dim - mean * mean;
    red[8] = mean;
    red[9] = rsqrtf(var + 1e-5f);
  }
  __syncthreads();
  const float mean = red[8], rstd = red[9];
  for (int j = threadIdx.x; j < dim; j += blockDim.x) {
    float v = (xr[j] - mean) * rstd * g[j] + bb[j];
    if (yf) yf[(size_t)row * dim + j] = v;
    if (yb) yb[(size_t)row * dim + j] = (bf16)v;
  }
}

// LayerNorm of z emitting BOTH layouts: zbf[(i,j)][c] and zT[i][c][j].
__global__ void layernorm_z_kernel(const float* __restrict__ z,
                                   const float* __restrict__ g,
                                   const float* __restrict__ bb,
                                   bf16* __restrict__ zbf,
                                   bf16* __restrict__ zT)
{
  __shared__ float red[10];
  const int row = blockIdx.x;               // row = i*NRES + j
  const int i = row / NRES, j = row % NRES;
  const float* xr = z + (size_t)row * CZ;
  float s = 0.f, s2 = 0.f;
  for (int c = threadIdx.x; c < CZ; c += blockDim.x) {
    float v = xr[c]; s += v; s2 += v * v;
  }
  for (int o = 16; o; o >>= 1) { s += __shfl_xor(s, o, 32); s2 += __shfl_xor(s2, o, 32); }
  if ((threadIdx.x & 31) == 0) { red[threadIdx.x >> 5] = s; red[4 + (threadIdx.x >> 5)] = s2; }
  __syncthreads();
  if (threadIdx.x == 0) {
    float a = 0.f, b2 = 0.f;
    const int nw = blockDim.x >> 5;
    for (int w = 0; w < nw; ++w) { a += red[w]; b2 += red[4 + w]; }
    const float mean = a / CZ;
    const float var  = b2 / CZ - mean * mean;
    red[8] = mean;
    red[9] = rsqrtf(var + 1e-5f);
  }
  __syncthreads();
  const float mean = red[8], rstd = red[9];
  for (int c = threadIdx.x; c < CZ; c += blockDim.x) {
    float v = (xr[c] - mean) * rstd * g[c] + bb[c];
    bf16 h = (bf16)v;
    zbf[(size_t)row * CZ + c] = h;
    zT[((size_t)i * CZ + c) * NRES + j] = h;
  }
}

__global__ void rigid_init_kernel(float* __restrict__ R, float* __restrict__ t)
{
  int i = blockIdx.x * blockDim.x + threadIdx.x;
  if (i >= NRES) return;
  float* Ri = R + i * 9;
  Ri[0] = 1.f; Ri[1] = 0.f; Ri[2] = 0.f;
  Ri[3] = 0.f; Ri[4] = 1.f; Ri[5] = 0.f;
  Ri[6] = 0.f; Ri[7] = 0.f; Ri[8] = 1.f;
  t[i * 3 + 0] = 0.f; t[i * 3 + 1] = 0.f; t[i * 3 + 2] = 0.f;
}

// q-points to world: qptsw[i][h][p][3] = R_i @ p_local + t_i ; raw [i][c*48+h*4+p]
__global__ void build_qpts_kernel(const bf16* __restrict__ raw,
                                  const float* __restrict__ R,
                                  const float* __restrict__ t,
                                  float* __restrict__ qptsw)
{
  int idx = blockIdx.x * blockDim.x + threadIdx.x;
  if (idx >= NRES * NH * NPQ) return;
  int i = idx / (NH * NPQ); int rem = idx % (NH * NPQ);
  int h = rem / NPQ, p = rem % NPQ;
  float lx = (float)raw[(size_t)i * 144 + 0 * 48 + h * 4 + p];
  float ly = (float)raw[(size_t)i * 144 + 1 * 48 + h * 4 + p];
  float lz = (float)raw[(size_t)i * 144 + 2 * 48 + h * 4 + p];
  const float* Ri = R + i * 9; const float* ti = t + i * 3;
  float* o = qptsw + ((size_t)i * NH + h) * 12 + p * 3;
  o[0] = Ri[0] * lx + Ri[1] * ly + Ri[2] * lz + ti[0];
  o[1] = Ri[3] * lx + Ri[4] * ly + Ri[5] * lz + ti[1];
  o[2] = Ri[6] * lx + Ri[7] * ly + Ri[8] * lz + ti[2];
}

// kv points: raw [i][c*144+h*12+p]; p<4 -> kptsw fp32; p>=4 -> vptsT bf16 (c-major)
__global__ void build_kvpts_kernel(const bf16* __restrict__ raw,
                                   const float* __restrict__ R,
                                   const float* __restrict__ t,
                                   float* __restrict__ kptsw,
                                   bf16* __restrict__ vptsT)
{
  int idx = blockIdx.x * blockDim.x + threadIdx.x;
  if (idx >= NRES * NH * 12) return;
  int i = idx / (NH * 12); int rem = idx % (NH * 12);
  int h = rem / 12, p = rem % 12;
  float lx = (float)raw[(size_t)i * 432 + 0 * 144 + h * 12 + p];
  float ly = (float)raw[(size_t)i * 432 + 1 * 144 + h * 12 + p];
  float lz = (float)raw[(size_t)i * 432 + 2 * 144 + h * 12 + p];
  const float* Ri = R + i * 9; const float* ti = t + i * 3;
  float wx = Ri[0] * lx + Ri[1] * ly + Ri[2] * lz + ti[0];
  float wy = Ri[3] * lx + Ri[4] * ly + Ri[5] * lz + ti[1];
  float wz = Ri[6] * lx + Ri[7] * ly + Ri[8] * lz + ti[2];
  if (p < NPQ) {
    float* o = kptsw + ((size_t)i * NH + h) * 12 + p * 3;
    o[0] = wx; o[1] = wy; o[2] = wz;
  } else {
    int d = (p - NPQ) * 3;
    vptsT[((size_t)h * 32 + d + 0) * NRES + i] = (bf16)wx;
    vptsT[((size_t)h * 32 + d + 1) * NRES + i] = (bf16)wy;
    vptsT[((size_t)h * 32 + d + 2) * NRES + i] = (bf16)wz;
  }
}

// v (from kv) -> c-major vT2 [h][16][NRES]
__global__ void transpose_v_kernel(const bf16* __restrict__ kvbf,
                                   bf16* __restrict__ vT2)
{
  int idx = blockIdx.x * blockDim.x + threadIdx.x;
  if (idx >= NRES * NH * CIPA) return;
  int i = idx / (NH * CIPA); int rem = idx % (NH * CIPA);
  int h = rem / CIPA, c = rem % CIPA;
  vT2[((size_t)h * CIPA + c) * NRES + i] =
      kvbf[(size_t)i * (NH * 2 * CIPA) + h * 32 + CIPA + c];
}

// Augmented q-hat / k-hat (K=32 zero-padded)
__global__ void build_hat_kernel(const bf16* __restrict__ qbf,
                                 const bf16* __restrict__ kvbf,
                                 const float* __restrict__ qptsw,
                                 const float* __restrict__ kptsw,
                                 const float* __restrict__ hwraw,
                                 bf16* __restrict__ qhat, bf16* __restrict__ khat,
                                 float* __restrict__ qn2, float* __restrict__ kn2)
{
  int idx = blockIdx.x * blockDim.x + threadIdx.x;
  if (idx >= NH * NRES) return;
  int h = idx / NRES, i = idx % NRES;
  const float hw = softplusf(hwraw[h]) * 0.13608276348795434f;  // sqrt(1/54)
  const float S1 = 0.14433756729740643f;                        // sqrt(1/48)
  bf16* qh = qhat + ((size_t)h * NRES + i) * 32;
  bf16* kh = khat + ((size_t)h * NRES + i) * 32;
  for (int c = 0; c < 16; ++c) {
    qh[c] = (bf16)((float)qbf[(size_t)i * (NH * CIPA) + h * CIPA + c] * S1);
    kh[c] = kvbf[(size_t)i * (NH * 2 * CIPA) + h * 32 + c];
  }
  float qs = 0.f, ks = 0.f;
  for (int d = 0; d < 12; ++d) {
    float qv = qptsw[((size_t)i * NH + h) * 12 + d];
    float kv = kptsw[((size_t)i * NH + h) * 12 + d];
    qh[16 + d] = (bf16)(qv * hw);
    kh[16 + d] = (bf16)kv;
    qs += qv * qv; ks += kv * kv;
  }
  for (int d = 28; d < 32; ++d) { qh[d] = (bf16)0.f; kh[d] = (bf16)0.f; }
  qn2[idx] = 0.5f * hw * qs;
  kn2[idx] = 0.5f * hw * ks;
}

// logits += sqrt(1/3)*biasT - (qn2+kn2) + INF*(m_i*m_j-1);  biasT [h][i][j]
__global__ void fixup_kernel(float* __restrict__ logits,
                             const float* __restrict__ biasT,
                             const float* __restrict__ qn2,
                             const float* __restrict__ kn2,
                             const float* __restrict__ mask)
{
  long idx = (long)blockIdx.x * blockDim.x + threadIdx.x;
  const long total = (long)NH * NRES * NRES;
  if (idx >= total) return;
  int h = (int)(idx / ((long)NRES * NRES));
  long rem = idx % ((long)NRES * NRES);
  int i = (int)(rem / NRES), j = (int)(rem % NRES);
  float v = logits[idx];
  v += 0.5773502691896258f * biasT[idx];
  v -= qn2[h * NRES + i] + kn2[h * NRES + j];
  v += 100000.f * (mask[i] * mask[j] - 1.f);
  logits[idx] = v;
}

// One wave per (h,i) row: softmax over 768 cols -> bf16 P
__global__ void softmax_kernel(const float* __restrict__ logits,
                               bf16* __restrict__ P)
{
  const int row  = blockIdx.x * (blockDim.x >> 5) + (threadIdx.x >> 5);
  const int lane = threadIdx.x & 31;
  const float* lr = logits + (size_t)row * NRES;
  float v[NRES / 32];
  float mx = -1e30f;
#pragma unroll
  for (int c = 0; c < NRES / 32; ++c) { v[c] = lr[c * 32 + lane]; mx = fmaxf(mx, v[c]); }
  for (int o = 16; o; o >>= 1) mx = fmaxf(mx, __shfl_xor(mx, o, 32));
  float sum = 0.f;
#pragma unroll
  for (int c = 0; c < NRES / 32; ++c) { v[c] = expf(v[c] - mx); sum += v[c]; }
  for (int o = 16; o; o >>= 1) sum += __shfl_xor(sum, o, 32);
  const float inv = 1.f / sum;
  bf16* pr = P + (size_t)row * NRES;
#pragma unroll
  for (int c = 0; c < NRES / 32; ++c) pr[c * 32 + lane] = (bf16)(v[c] * inv);
}

// Assemble 2112-wide concat row (bf16).
// obuf [h][N][16]; optbuf [h][N][32]; opairb [i][128][16] (c-major, h minor)
__global__ void cat_kernel(const float* __restrict__ obuf,
                           const float* __restrict__ optbuf,
                           const float* __restrict__ opairb,
                           const float* __restrict__ R,
                           const float* __restrict__ t,
                           bf16* __restrict__ cat)
{
  const int i = blockIdx.x;
  __shared__ float lx[96], ly[96], lz[96], lnorm[96];
  const float* Ri = R + i * 9;
  const float tx = t[i * 3], ty = t[i * 3 + 1], tz = t[i * 3 + 2];
  for (int idx = threadIdx.x; idx < 96; idx += blockDim.x) {
    int h = idx / NPV, p = idx % NPV;
    const float* op = optbuf + ((size_t)h * NRES + i) * 32 + p * 3;
    float wx = op[0] - tx, wy = op[1] - ty, wz = op[2] - tz;
    float x = Ri[0] * wx + Ri[3] * wy + Ri[6] * wz;   // R^T
    float y = Ri[1] * wx + Ri[4] * wy + Ri[7] * wz;
    float z = Ri[2] * wx + Ri[5] * wy + Ri[8] * wz;
    lx[idx] = x; ly[idx] = y; lz[idx] = z;
    lnorm[idx] = sqrtf(x * x + y * y + z * z + 1e-8f);
  }
  __syncthreads();
  bf16* cr = cat + (size_t)i * CATD;
  for (int j = threadIdx.x; j < CATD; j += blockDim.x) {
    float v;
    if (j < 192)       { int h = j >> 4, c = j & 15; v = obuf[((size_t)h * NRES + i) * 16 + c]; }
    else if (j < 288)  v = lx[j - 192];
    else if (j < 384)  v = ly[j - 288];
    else if (j < 480)  v = lz[j - 384];
    else if (j < 576)  v = lnorm[j - 480];
    else { int k = j - 576; int h = k >> 7, c = k & 127;
           v = opairb[((size_t)i * 128 + c) * 16 + h]; }
    cr[j] = (bf16)v;
  }
}

// Backbone update
__global__ void rigid_update_kernel(const float* __restrict__ upd,
                                    float* __restrict__ R, float* __restrict__ t)
{
  int i = blockIdx.x * blockDim.x + threadIdx.x;
  if (i >= NRES) return;
  const float* u = upd + i * 6;
  float qw = 1.f, qx = u[0], qy = u[1], qz = u[2];
  float inv = rsqrtf(qw * qw + qx * qx + qy * qy + qz * qz);
  qw *= inv; qx *= inv; qy *= inv; qz *= inv;
  float Ru[9] = {
    1.f - 2.f * (qy * qy + qz * qz), 2.f * (qx * qy - qw * qz), 2.f * (qx * qz + qw * qy),
    2.f * (qx * qy + qw * qz), 1.f - 2.f * (qx * qx + qz * qz), 2.f * (qy * qz - qw * qx),
    2.f * (qx * qz - qw * qy), 2.f * (qy * qz + qw * qx), 1.f - 2.f * (qx * qx + qy * qy)};
  float* Ri = R + i * 9; float* ti = t + i * 3;
  float Ro[9];
#pragma unroll
  for (int k = 0; k < 9; ++k) Ro[k] = Ri[k];
#pragma unroll
  for (int r = 0; r < 3; ++r)
    ti[r] += Ro[r * 3 + 0] * u[3] + Ro[r * 3 + 1] * u[4] + Ro[r * 3 + 2] * u[5];
#pragma unroll
  for (int r = 0; r < 3; ++r)
#pragma unroll
    for (int c = 0; c < 3; ++c)
      Ri[r * 3 + c] = Ro[r * 3 + 0] * Ru[0 * 3 + c] + Ro[r * 3 + 1] * Ru[1 * 3 + c]
                    + Ro[r * 3 + 2] * Ru[2 * 3 + c];
}

// Final output: [s(384) | unnorm(14) | angles(14) | 10*t(3) | R(9)]
__global__ void finalize_kernel(const float* __restrict__ s,
                                const float* __restrict__ ang,
                                const float* __restrict__ R,
                                const float* __restrict__ t,
                                float* __restrict__ out)
{
  const int i = blockIdx.x;
  float* o = out + (size_t)i * 424;
  for (int j = threadIdx.x; j < 424; j += blockDim.x) {
    float v;
    if (j < 384) v = s[(size_t)i * 384 + j];
    else if (j < 398) v = ang[i * 14 + (j - 384)];
    else if (j < 412) {
      int k = j - 398; int a = k >> 1;
      float a0 = ang[i * 14 + 2 * a], a1 = ang[i * 14 + 2 * a + 1];
      float ss = a0 * a0 + a1 * a1; ss = ss < 1e-8f ? 1e-8f : ss;
      v = ang[i * 14 + k] * rsqrtf(ss);
    }
    else if (j < 415) v = t[i * 3 + (j - 412)] * 10.f;
    else v = R[i * 9 + (j - 415)];
    o[j] = v;
  }
}

// ===========================================================================
// Host orchestration
// ===========================================================================
static inline void gemmF(hipStream_t st,
                         const bf16* A, int lda, size_t sA,
                         const bf16* B, int ldb, size_t sB,
                         const float* bias, int biasRow,
                         const float* Res, int ldr,
                         float* Cf, int ldcf, size_t sCf,
                         bf16* Cb, int ldcb, size_t sCb,
                         int M, int N, int K, int relu, int batch)
{
  dim3 g((N + 31) / 32, (M + 127) / 128, batch);
  wmma_gemm_fast_kernel<<<g, 128, 0, st>>>(A, lda, sA, B, ldb, sB, bias, biasRow,
                                           Res, ldr, Cf, ldcf, sCf, Cb, ldcb, sCb,
                                           M, N, K, relu);
}

static inline void gemmE(hipStream_t st,
                         const bf16* A, int lda, const bf16* B, int ldb,
                         const float* bias, const float* Res, int ldr,
                         float* Cf, int ldcf, int M, int N, int K)
{
  dim3 g((N + 15) / 16, (M + 63) / 64, 1);
  wmma_gemm_edge_kernel<<<g, 128, 0, st>>>(A, lda, B, ldb, bias, Res, ldr,
                                           Cf, ldcf, M, N, K);
}

static inline void cvt(hipStream_t st, const float* in, bf16* out, int n, int relu)
{
  cvt_kernel<<<(n + 255) / 256, 256, 0, st>>>(in, out, n, relu);
}

static inline void transw(hipStream_t st, const float* W, bf16* WT, int K, int N)
{
  transpose_w_kernel<<<(K * N + 255) / 256, 256, 0, st>>>(W, WT, K, N);
}

extern "C" void kernel_launch(void* const* d_in, const int* in_sizes, int n_in,
                              void* d_out, int out_size, void* d_ws, size_t ws_size,
                              hipStream_t stream)
{
  const float* s_in   = (const float*)d_in[0];
  const float* z_in   = (const float*)d_in[1];
  const float* mask   = (const float*)d_in[2];
  const float* lns_g  = (const float*)d_in[3];
  const float* lns_b  = (const float*)d_in[4];
  const float* lnz_g  = (const float*)d_in[5];
  const float* lnz_b  = (const float*)d_in[6];
  const float* Wlin   = (const float*)d_in[7];
  const float* blin   = (const float*)d_in[8];
  const float* Wq     = (const float*)d_in[9];
  const float* bq     = (const float*)d_in[10];
  const float* Wkv    = (const float*)d_in[11];
  const float* bkv    = (const float*)d_in[12];
  const float* Wqp    = (const float*)d_in[13];
  const float* bqp    = (const float*)d_in[14];
  const float* Wkvp   = (const float*)d_in[15];
  const float* bkvp   = (const float*)d_in[16];
  const float* Wpb    = (const float*)d_in[17];
  const float* bpb    = (const float*)d_in[18];
  const float* hwraw  = (const float*)d_in[19];
  const float* Wout   = (const float*)d_in[20];
  const float* bout   = (const float*)d_in[21];
  const float* lni_g  = (const float*)d_in[22];
  const float* lni_b  = (const float*)d_in[23];
  const float* Wt1    = (const float*)d_in[24];
  const float* bt1    = (const float*)d_in[25];
  const float* Wt2    = (const float*)d_in[26];
  const float* bt2    = (const float*)d_in[27];
  const float* Wt3    = (const float*)d_in[28];
  const float* bt3    = (const float*)d_in[29];
  const float* lnt_g  = (const float*)d_in[30];
  const float* lnt_b  = (const float*)d_in[31];
  const float* Wbb    = (const float*)d_in[32];
  const float* bbb    = (const float*)d_in[33];
  const float* Wai    = (const float*)d_in[34];  // lin_initial
  const float* bai    = (const float*)d_in[35];
  const float* Wan    = (const float*)d_in[36];  // lin_in
  const float* ban    = (const float*)d_in[37];
  const float* Wb0l2  = (const float*)d_in[38];
  const float* bb0l2  = (const float*)d_in[39];
  const float* Wb0l3  = (const float*)d_in[40];
  const float* bb0l3  = (const float*)d_in[41];
  const float* Wb1l2  = (const float*)d_in[42];
  const float* bb1l2  = (const float*)d_in[43];
  const float* Wb1l3  = (const float*)d_in[44];
  const float* bb1l3  = (const float*)d_in[45];
  const float* Wao    = (const float*)d_in[46];
  const float* bao    = (const float*)d_in[47];
  float* outp = (float*)d_out;

  char* base = (char*)d_ws;
  size_t off = 0;
  auto alloc = [&](size_t bytes) -> void* {
    void* p = base + off;
    off += (bytes + 255) & ~(size_t)255;
    return p;
  };
  const size_t NN = (size_t)NRES * NRES;

  float* sln    = (float*)alloc(sizeof(float) * NRES * CS);
  float* scur   = (float*)alloc(sizeof(float) * NRES * CS);
  bf16*  sbf    = (bf16*) alloc(sizeof(bf16)  * NRES * CS);
  bf16*  t1bf   = (bf16*) alloc(sizeof(bf16)  * NRES * CS);
  bf16*  t2bf   = (bf16*) alloc(sizeof(bf16)  * NRES * CS);
  bf16*  zbf    = (bf16*) alloc(sizeof(bf16)  * NN * CZ);        // [ij][c]
  bf16*  zT     = (bf16*) alloc(sizeof(bf16)  * NN * CZ);        // [i][c][j]
  float* biasT  = (float*)alloc(sizeof(float) * 16 * NN);        // [h][i][j], 16 heads padded
  bf16*  qbf    = (bf16*) alloc(sizeof(bf16)  * NRES * NH * CIPA);
  bf16*  kvbf   = (bf16*) alloc(sizeof(bf16)  * NRES * NH * 2 * CIPA);
  bf16*  qpraw  = (bf16*) alloc(sizeof(bf16)  * NRES * 144);
  bf16*  kvpraw = (bf16*) alloc(sizeof(bf16)  * NRES * 432);
  float* qptsw  = (float*)alloc(sizeof(float) * NRES * NH * 12);
  float* kptsw  = (float*)alloc(sizeof(float) * NRES * NH * 12);
  bf16*  vptsT  = (bf16*) alloc(sizeof(bf16)  * NH * 32 * NRES); // padded 24->32
  bf16*  vT2    = (bf16*) alloc(sizeof(bf16)  * NH * CIPA * NRES);
  bf16*  qhat   = (bf16*) alloc(sizeof(bf16)  * NH * NRES * 32);
  bf16*  khat   = (bf16*) alloc(sizeof(bf16)  * NH * NRES * 32);
  float* qn2    = (float*)alloc(sizeof(float) * NH * NRES);
  float* kn2    = (float*)alloc(sizeof(float) * NH * NRES);
  float* logits = (float*)alloc(sizeof(float) * NH * NN);
  bf16*  Pbf    = (bf16*) alloc(sizeof(bf16)  * 16 * NN);        // padded 12->16 heads
  float* obuf   = (float*)alloc(sizeof(float) * NH * NRES * 16);
  float* optbuf = (float*)alloc(sizeof(float) * NH * NRES * 32);
  float* opairb = (float*)alloc(sizeof(float) * NRES * 128 * 16);
  bf16*  catbf  = (bf16*) alloc(sizeof(bf16)  * NRES * CATD);
  float* updb   = (float*)alloc(sizeof(float) * NRES * 6);
  float* Rbuf   = (float*)alloc(sizeof(float) * NRES * 9);
  float* tbuf   = (float*)alloc(sizeof(float) * NRES * 3);
  float* abuf   = (float*)alloc(sizeof(float) * NRES * 128);
  bf16*  abf    = (bf16*) alloc(sizeof(bf16)  * NRES * 128);
  bf16*  h1bf   = (bf16*) alloc(sizeof(bf16)  * NRES * 128);
  float* angb   = (float*)alloc(sizeof(float) * NRES * 14);
  float* bpb16  = (float*)alloc(sizeof(float) * 16);

  // transposed bf16 weights [n][k]
  bf16* WqT   = (bf16*)alloc(sizeof(bf16) * 192 * CS);
  bf16* WkvT  = (bf16*)alloc(sizeof(bf16) * 384 * CS);
  bf16* WqpT  = (bf16*)alloc(sizeof(bf16) * 144 * CS);
  bf16* WkvpT = (bf16*)alloc(sizeof(bf16) * 432 * CS);
  bf16* WpbT  = (bf16*)alloc(sizeof(bf16) * 16 * CZ);   // padded 12->16 rows
  bf16* WoutT = (bf16*)alloc(sizeof(bf16) * CS * CATD);
  bf16* WlinT = (bf16*)alloc(sizeof(bf16) * CS * CS);
  bf16* Wt1T  = (bf16*)alloc(sizeof(bf16) * CS * CS);
  bf16* Wt2T  = (bf16*)alloc(sizeof(bf16) * CS * CS);
  bf16* Wt3T  = (bf16*)alloc(sizeof(bf16) * CS * CS);
  bf16* WbbT  = (bf16*)alloc(sizeof(bf16) * 6 * CS);
  bf16* WaiT  = (bf16*)alloc(sizeof(bf16) * 128 * CS);
  bf16* WanT  = (bf16*)alloc(sizeof(bf16) * 128 * CS);
  bf16* Wb0l2T = (bf16*)alloc(sizeof(bf16) * 128 * 128);
  bf16* Wb0l3T = (bf16*)alloc(sizeof(bf16) * 128 * 128);
  bf16* Wb1l2T = (bf16*)alloc(sizeof(bf16) * 128 * 128);
  bf16* Wb1l3T = (bf16*)alloc(sizeof(bf16) * 128 * 128);
  bf16* WaoT  = (bf16*)alloc(sizeof(bf16) * 14 * 128);

  // ---- setup: weight transposes + padding ----
  transw(stream, Wq,   WqT,   CS, 192);
  transw(stream, Wkv,  WkvT,  CS, 384);
  transw(stream, Wqp,  WqpT,  CS, 144);
  transw(stream, Wkvp, WkvpT, CS, 432);
  transw(stream, Wpb,  WpbT,  CZ, NH);
  transw(stream, Wout, WoutT, CATD, CS);
  transw(stream, Wlin, WlinT, CS, CS);
  transw(stream, Wt1,  Wt1T,  CS, CS);
  transw(stream, Wt2,  Wt2T,  CS, CS);
  transw(stream, Wt3,  Wt3T,  CS, CS);
  transw(stream, Wbb,  WbbT,  CS, 6);
  transw(stream, Wai,  WaiT,  CS, 128);
  transw(stream, Wan,  WanT,  CS, 128);
  transw(stream, Wb0l2, Wb0l2T, 128, 128);
  transw(stream, Wb0l3, Wb0l3T, 128, 128);
  transw(stream, Wb1l2, Wb1l2T, 128, 128);
  transw(stream, Wb1l3, Wb1l3T, 128, 128);
  transw(stream, Wao,  WaoT,  128, 14);
  zero_bf16_kernel<<<(4 * CZ + 255) / 256, 256, 0, stream>>>(WpbT + NH * CZ, 4 * CZ);
  zero_bf16_kernel<<<(int)((4 * NN + 255) / 256), 256, 0, stream>>>(
      Pbf + 12 * NN, (int)(4 * NN));
  zero_vptst_pad_kernel<<<(NH * 8 * NRES + 255) / 256, 256, 0, stream>>>(vptsT);
  pad_bias_kernel<<<1, 16, 0, stream>>>(bpb, bpb16, NH, 16);

  // ---- LayerNorms; z in both layouts (bf16, L2-resident) ----
  layernorm_kernel<<<NRES, 128, 0, stream>>>(s_in, lns_g, lns_b, sln, nullptr, CS);
  layernorm_z_kernel<<<(int)NN, 128, 0, stream>>>(z_in, lnz_g, lnz_b, zbf, zT);

  // ---- hoisted pair bias: biasT[h][(ij)] = Wb^T @ z^T + b  (M padded to 16) ----
  gemmF(stream, WpbT, CZ, 0, zbf, CZ, 0, bpb16, 1, nullptr, 0,
        biasT, (int)NN, 0, nullptr, 0, 0, 16, (int)NN, CZ, 0, 1);

  // ---- s = LN(s) @ W_in + b ----
  cvt(stream, sln, sbf, NRES * CS, 0);
  gemmF(stream, sbf, CS, 0, WlinT, CS, 0, blin, 0, nullptr, 0,
        scur, CS, 0, nullptr, 0, 0, NRES, CS, CS, 0, 1);

  rigid_init_kernel<<<(NRES + 255) / 256, 256, 0, stream>>>(Rbuf, tbuf);

  for (int blk = 0; blk < NBLOCKS; ++blk) {
    cvt(stream, scur, sbf, NRES * CS, 0);
    gemmF(stream, sbf, CS, 0, WqT, CS, 0, bq, 0, nullptr, 0,
          nullptr, 0, 0, qbf, 192, 0, NRES, 192, CS, 0, 1);
    gemmF(stream, sbf, CS, 0, WkvT, CS, 0, bkv, 0, nullptr, 0,
          nullptr, 0, 0, kvbf, 384, 0, NRES, 384, CS, 0, 1);
    gemmF(stream, sbf, CS, 0, WqpT, CS, 0, bqp, 0, nullptr, 0,
          nullptr, 0, 0, qpraw, 144, 0, NRES, 144, CS, 0, 1);
    gemmF(stream, sbf, CS, 0, WkvpT, CS, 0, bkvp, 0, nullptr, 0,
          nullptr, 0, 0, kvpraw, 432, 0, NRES, 432, CS, 0, 1);

    build_qpts_kernel<<<(NRES * NH * NPQ + 255) / 256, 256, 0, stream>>>(
        qpraw, Rbuf, tbuf, qptsw);
    build_kvpts_kernel<<<(NRES * NH * 12 + 255) / 256, 256, 0, stream>>>(
        kvpraw, Rbuf, tbuf, kptsw, vptsT);
    transpose_v_kernel<<<(NRES * NH * CIPA + 255) / 256, 256, 0, stream>>>(kvbf, vT2);
    build_hat_kernel<<<(NH * NRES + 255) / 256, 256, 0, stream>>>(
        qbf, kvbf, qptsw, kptsw, hwraw, qhat, khat, qn2, kn2);

    // logits[h] = qhat @ khat^T   (K=32, point attention folded in)
    gemmF(stream, qhat, 32, (size_t)NRES * 32, khat, 32, (size_t)NRES * 32,
          nullptr, 0, nullptr, 0, logits, NRES, NN, nullptr, 0, 0,
          NRES, NRES, 32, 0, NH);
    fixup_kernel<<<(int)((NH * NN + 255) / 256), 256, 0, stream>>>(
        logits, biasT, qn2, kn2, mask);
    softmax_kernel<<<NH * NRES / 4, 128, 0, stream>>>(logits, Pbf);

    // o = P @ v ; o_pt = P @ v_pts  (per-head, NT-fast B)
    gemmF(stream, Pbf, NRES, NN, vT2, NRES, (size_t)CIPA * NRES,
          nullptr, 0, nullptr, 0, obuf, 16, (size_t)NRES * 16, nullptr, 0, 0,
          NRES, 16, NRES, 0, NH);
    gemmF(stream, Pbf, NRES, NN, vptsT, NRES, (size_t)32 * NRES,
          nullptr, 0, nullptr, 0, optbuf, 32, (size_t)NRES * 32, nullptr, 0, 0,
          NRES, 32, NRES, 0, NH);
    // o_pair^T_i = z^T_i (128x768) @ P[:, i, :]^T (batched over i; 16-head pad)
    gemmF(stream, zT, NRES, (size_t)CZ * NRES, Pbf, (int)NN, NRES,
          nullptr, 0, nullptr, 0, opairb, 16, (size_t)CZ * 16, nullptr, 0, 0,
          CZ, 16, NRES, 0, NRES);

    cat_kernel<<<NRES, 128, 0, stream>>>(obuf, optbuf, opairb, Rbuf, tbuf, catbf);

    gemmF(stream, catbf, CATD, 0, WoutT, CATD, 0, bout, 0, scur, CS,
          scur, CS, 0, nullptr, 0, 0, NRES, CS, CATD, 0, 1);
    layernorm_kernel<<<NRES, 128, 0, stream>>>(scur, lni_g, lni_b, scur, nullptr, CS);

    cvt(stream, scur, sbf, NRES * CS, 0);
    gemmF(stream, sbf, CS, 0, Wt1T, CS, 0, bt1, 0, nullptr, 0,
          nullptr, 0, 0, t1bf, CS, 0, NRES, CS, CS, 1, 1);
    gemmF(stream, t1bf, CS, 0, Wt2T, CS, 0, bt2, 0, nullptr, 0,
          nullptr, 0, 0, t2bf, CS, 0, NRES, CS, CS, 1, 1);
    gemmF(stream, t2bf, CS, 0, Wt3T, CS, 0, bt3, 0, scur, CS,
          scur, CS, 0, nullptr, 0, 0, NRES, CS, CS, 0, 1);
    layernorm_kernel<<<NRES, 128, 0, stream>>>(scur, lnt_g, lnt_b, scur, nullptr, CS);

    cvt(stream, scur, sbf, NRES * CS, 0);
    gemmE(stream, sbf, CS, WbbT, CS, bbb, nullptr, 0, updb, 6, NRES, 6, CS);
    rigid_update_kernel<<<(NRES + 255) / 256, 256, 0, stream>>>(updb, Rbuf, tbuf);
  }

  // ---- angle resnet ----
  cvt(stream, scur, sbf, NRES * CS, 1);    // relu(s)
  cvt(stream, sln,  t1bf, NRES * CS, 1);   // relu(s_initial)
  gemmF(stream, t1bf, CS, 0, WaiT, CS, 0, bai, 0, nullptr, 0,
        abuf, 128, 0, nullptr, 0, 0, NRES, 128, CS, 0, 1);
  gemmF(stream, sbf, CS, 0, WanT, CS, 0, ban, 0, abuf, 128,
        abuf, 128, 0, nullptr, 0, 0, NRES, 128, CS, 0, 1);

  const bf16* blkW2[2] = {Wb0l2T, Wb1l2T};
  const bf16* blkW3[2] = {Wb0l3T, Wb1l3T};
  const float* blkb2[2] = {bb0l2, bb1l2};
  const float* blkb3[2] = {bb0l3, bb1l3};
  for (int r = 0; r < 2; ++r) {
    cvt(stream, abuf, abf, NRES * 128, 1);
    gemmF(stream, abf, 128, 0, blkW2[r], 128, 0, blkb2[r], 0, nullptr, 0,
          nullptr, 0, 0, h1bf, 128, 0, NRES, 128, 128, 1, 1);
    gemmF(stream, h1bf, 128, 0, blkW3[r], 128, 0, blkb3[r], 0, abuf, 128,
          abuf, 128, 0, nullptr, 0, 0, NRES, 128, 128, 0, 1);
  }
  cvt(stream, abuf, abf, NRES * 128, 1);
  gemmE(stream, abf, 128, WaoT, 128, bao, nullptr, 0, angb, 14, NRES, 14, 128);

  finalize_kernel<<<NRES, 128, 0, stream>>>(scur, angb, Rbuf, tbuf, outp);
}
